// MultiHeadAttention_86629490360598
// MI455X (gfx1250) — matrix operand order repaired and checked by reading, not compiled, not run on hardware
//
#include <hip/hip_runtime.h>

// ---------------------------------------------------------------------------
// MHA forward for gfx1250 (MI455X): flash attention + bf16 WMMA GEMMs with
// register blocking for operand reuse.
//
// WMMA fragment layouts per CDNA5 ISA 7.12.2 (16x16x32 bf16):
//   A (16x32): lane&15 = row M, hi=lane>>4; elem j<8: K=hi*8+j ; j>=8: K=16+hi*8+(j-8)
//   B (32x16): lane&15 = col N, hi=lane>>4; elem j: K as for A (this code loads both A and B with K=hi*16+j)
//   C/D (16x16 f32): lane&15 = col N, hi=lane>>4; reg r: row M = r + 8*hi
// ---------------------------------------------------------------------------

typedef __attribute__((ext_vector_type(16))) __bf16 v16bf;
typedef __attribute__((ext_vector_type(8)))  __bf16 v8bf;
typedef __attribute__((ext_vector_type(8)))  float  v8f;

#define BATCH   4
#define SEQ     2048
#define DMODEL  1024
#define NHEADS  16
#define DHEAD   64
#define BS      (BATCH * SEQ)          // 8192 flattened rows
#define BH      (BATCH * NHEADS)       // 64   (b,h) pairs
#define QSCALE  0.125f                 // 1/sqrt(64)
#define MASKV   (-1.0e6f)

static __device__ __forceinline__ __bf16 f2bf(float f) {
  unsigned u = __builtin_bit_cast(unsigned, f);
  unsigned r = u + 0x7FFFu + ((u >> 16) & 1u);   // round-to-nearest-even
  unsigned short h = (unsigned short)(r >> 16);
  return __builtin_bit_cast(__bf16, h);
}

static __device__ __forceinline__ v8f vzero8() {
  v8f z;
#pragma unroll
  for (int i = 0; i < 8; ++i) z[i] = 0.0f;
  return z;
}

// A fragment: 16x32 bf16 tile from row-major [.., ld] source at (row0, k0).
static __device__ __forceinline__ v16bf load_afrag(const __bf16* base, int ld,
                                                   int row0, int k0, int lane) {
  int m = lane & 15, hi = lane >> 4;
  const __bf16* p = base + (size_t)(row0 + m) * ld + k0 + hi * 16;
  v8bf lo = *reinterpret_cast<const v8bf*>(p);
  v8bf hh = *reinterpret_cast<const v8bf*>(p + 8);
  v16bf a;
#pragma unroll
  for (int i = 0; i < 8; ++i) { a[i] = lo[i]; a[i + 8] = hh[i]; }
  return a;
}

// B fragment: 32x16 bf16 tile, element (k,n) = base[(n0+n)*ld + k0 + k].
static __device__ __forceinline__ v16bf load_bfrag(const __bf16* base, int ld,
                                                   int n0, int k0, int lane) {
  int n = lane & 15, hi = lane >> 4;
  const __bf16* p = base + (size_t)(n0 + n) * ld + k0 + hi * 16;
  v8bf lo = *reinterpret_cast<const v8bf*>(p);
  v8bf hh = *reinterpret_cast<const v8bf*>(p + 8);
  v16bf b;
#pragma unroll
  for (int i = 0; i < 8; ++i) { b[i] = lo[i]; b[i + 8] = hh[i]; }
  return b;
}

static __device__ __forceinline__ v8f wmma_bf16(v16bf a, v16bf b, v8f c) {
  return __builtin_amdgcn_wmma_f32_16x16x32_bf16(false, a, false, b,
                                                 (short)0, c, false, false);
}

// ---------------------------------------------------------------------------
// Prep kernels
// ---------------------------------------------------------------------------

__global__ __launch_bounds__(256) void k_conv_bf16(const float* __restrict__ src,
                                                   __bf16* __restrict__ dst,
                                                   int n) {
  int stride = gridDim.x * blockDim.x;
  for (int i = blockIdx.x * blockDim.x + threadIdx.x; i < n; i += stride)
    dst[i] = f2bf(src[i]);
}

// dst[oc*1024 + dd] = src[dd*1024 + oc]   (W_Q/K/V: [d_model][h*dh] -> T)
__global__ __launch_bounds__(256) void k_transpose_w(const float* __restrict__ src,
                                                     __bf16* __restrict__ dst) {
  int i = blockIdx.x * blockDim.x + threadIdx.x;   // 1M elements exactly
  int oc = i >> 10, dd = i & 1023;
  dst[(size_t)oc * 1024 + dd] = f2bf(src[(size_t)dd * 1024 + oc]);
}

// W_out [c][h][d] -> WoutT [d][h*64+c]
__global__ __launch_bounds__(256) void k_prep_wout(const float* __restrict__ src,
                                                   __bf16* __restrict__ dst) {
  int i = blockIdx.x * blockDim.x + threadIdx.x;   // 1M elements exactly
  int d = i >> 10, oc = i & 1023;
  int h = oc >> 6, c = oc & 63;
  dst[i] = f2bf(src[((size_t)c * 16 + h) * 1024 + d]);
}

// ---------------------------------------------------------------------------
// QKV projection: [8192,1024] x [1024,1024] per matrix.
// Register-blocked: each wave computes a 64x32 output block (4x2 WMMA tiles).
// Writes q (scaled, [bh][s][64]), k ([bh][s][64]), vT ([bh][64][s]) in bf16.
// ---------------------------------------------------------------------------
__global__ __launch_bounds__(256) void k_qkv(const __bf16* __restrict__ resid_bf,
                                             const __bf16* __restrict__ wqt,
                                             const __bf16* __restrict__ wkt,
                                             const __bf16* __restrict__ wvt,
                                             __bf16* __restrict__ q_all,
                                             __bf16* __restrict__ k_all,
                                             __bf16* __restrict__ vT_all) {
  int wave = blockIdx.x * 8 + (threadIdx.x >> 5);
  int lane = threadIdx.x & 31;
  // 128 row-tiles(64) x 32 col-tiles(32) per matrix, 3 matrices
  int mat = wave / (128 * 32);
  int t   = wave % (128 * 32);
  int rt = t >> 5, ct = t & 31;
  const __bf16* wt = (mat == 0) ? wqt : (mat == 1) ? wkt : wvt;

  v8f acc[4][2];
#pragma unroll
  for (int i = 0; i < 4; ++i)
#pragma unroll
    for (int j = 0; j < 2; ++j) acc[i][j] = vzero8();

#pragma unroll 2
  for (int k0 = 0; k0 < DMODEL; k0 += 32) {
    v16bf a[4], b[2];
#pragma unroll
    for (int i = 0; i < 4; ++i)
      a[i] = load_afrag(resid_bf, DMODEL, rt * 64 + i * 16, k0, lane);
#pragma unroll
    for (int j = 0; j < 2; ++j)
      b[j] = load_bfrag(wt, DMODEL, ct * 32 + j * 16, k0, lane);
#pragma unroll
    for (int i = 0; i < 4; ++i)
#pragma unroll
      for (int j = 0; j < 2; ++j) acc[i][j] = wmma_bf16(a[i], b[j], acc[i][j]);
  }

  int hi = lane >> 4, n = lane & 15;
#pragma unroll
  for (int i = 0; i < 4; ++i)
#pragma unroll
    for (int j = 0; j < 2; ++j) {
      int col = ct * 32 + j * 16 + n;
      int h = col >> 6, cc = col & 63;
#pragma unroll
      for (int r = 0; r < 8; ++r) {
        int row = rt * 64 + i * 16 + hi * 8 + r;
        int b_ = row >> 11, s = row & 2047;
        size_t bh = (size_t)b_ * NHEADS + h;
        float v = acc[i][j][r];
        if (mat == 0)      q_all[(bh * SEQ + s) * DHEAD + cc] = f2bf(v * QSCALE);
        else if (mat == 1) k_all[(bh * SEQ + s) * DHEAD + cc] = f2bf(v);
        else               vT_all[(bh * DHEAD + cc) * SEQ + s] = f2bf(v);
      }
    }
}

// ---------------------------------------------------------------------------
// Flash attention. One wave per (b,h, 32-query block): two query-column tiles
// share the K/V fragments. Scores computed transposed (Sᵀ = K·Qᵀ) so each
// lane owns a query column; O accumulated transposed (Oᵀ = Vᵀ·P).
// ---------------------------------------------------------------------------

// Online-softmax update for one query tile over a 32-key chunk.
// s0/s1: Sᵀ tiles (keys key0..+15 / +16..+31, this lane's query column).
// Returns the P fragment (B-frag: 32 keys x 16 queries) for the PV WMMAs.
static __device__ __forceinline__ v16bf softmax_update(v8f s0, v8f s1,
                                                       float& m_i, float& l_i,
                                                       v8f og[4],
                                                       int hi, int qg, int key0) {
  // causal mask: reg r of tile t holds key = key0 + t*16 + r + 8*hi
#pragma unroll
  for (int r = 0; r < 8; ++r) {
    if (key0 + r + 8 * hi > qg)      s0[r] = MASKV;
    if (key0 + 16 + r + 8 * hi > qg) s1[r] = MASKV;
  }
  float mc = -1.0e30f;
#pragma unroll
  for (int r = 0; r < 8; ++r) mc = fmaxf(mc, fmaxf(s0[r], s1[r]));
  mc = fmaxf(mc, __shfl_xor(mc, 16, 32));
  float m_new = fmaxf(m_i, mc);
  float alpha = __expf(m_i - m_new);
  float p0[8], p1[8], lsum = 0.0f;
#pragma unroll
  for (int r = 0; r < 8; ++r) {
    p0[r] = __expf(s0[r] - m_new);
    p1[r] = __expf(s1[r] - m_new);
    lsum += p0[r] + p1[r];
  }
  lsum += __shfl_xor(lsum, 16, 32);
  l_i = l_i * alpha + lsum;
  m_i = m_new;
#pragma unroll
  for (int g = 0; g < 4; ++g)
#pragma unroll
    for (int r = 0; r < 8; ++r) og[g][r] *= alpha;
  // assemble P as B fragment via xor-16 shuffles
  v16bf pb;
#pragma unroll
  for (int r = 0; r < 8; ++r) {
    float o0 = __shfl_xor(p0[r], 16, 32);
    float o1 = __shfl_xor(p1[r], 16, 32);
    pb[r]     = hi ? f2bf(o1)    : f2bf(p0[r]);   // elem j=r   -> K=hi*16+r
    pb[r + 8] = hi ? f2bf(p1[r]) : f2bf(o0);      // elem j=r+8 -> K=hi*16+8+r
  }
  return pb;
}

__global__ __launch_bounds__(256) void k_attn(const __bf16* __restrict__ q_all,
                                              const __bf16* __restrict__ k_all,
                                              const __bf16* __restrict__ vT_all,
                                              __bf16* __restrict__ z_all) {
  int wid  = blockIdx.x * 8 + (threadIdx.x >> 5);   // 0..4095
  int lane = threadIdx.x & 31;
  int qt = wid & 63;                  // 32-query block within (b,h)
  int bh = wid >> 6;                  // 0..63
  const __bf16* qb = q_all  + (size_t)bh * SEQ * DHEAD;
  const __bf16* kb = k_all  + (size_t)bh * SEQ * DHEAD;
  const __bf16* vb = vT_all + (size_t)bh * DHEAD * SEQ;

  int q0 = qt * 32;
  int hi = lane >> 4, nq = lane & 15;
  int qgA = q0 + nq;                  // query tile A column
  int qgB = q0 + 16 + nq;             // query tile B column

  float mA = -1.0e30f, lA = 0.0f, mB = -1.0e30f, lB = 0.0f;
  v8f ogA[4], ogB[4];
#pragma unroll
  for (int g = 0; g < 4; ++g) { ogA[g] = vzero8(); ogB[g] = vzero8(); }

  int nchunks = qt + 1;               // causal: keys 0 .. q0+31
  for (int ch = 0; ch < nchunks; ++ch) {
    int key0 = ch * 32;
    // ---- Sᵀ tiles: K A-frags shared by both Q B-frags
    v8f sA0 = vzero8(), sA1 = vzero8(), sB0 = vzero8(), sB1 = vzero8();
#pragma unroll
    for (int c0 = 0; c0 < DHEAD; c0 += 32) {
      v16bf a0  = load_afrag(kb, DHEAD, key0,      c0, lane);
      v16bf a1  = load_afrag(kb, DHEAD, key0 + 16, c0, lane);
      v16bf bq0 = load_bfrag(qb, DHEAD, q0,        c0, lane);
      v16bf bq1 = load_bfrag(qb, DHEAD, q0 + 16,   c0, lane);
      sA0 = wmma_bf16(a0, bq0, sA0);
      sA1 = wmma_bf16(a1, bq0, sA1);
      sB0 = wmma_bf16(a0, bq1, sB0);
      sB1 = wmma_bf16(a1, bq1, sB1);
    }
    v16bf pbA = softmax_update(sA0, sA1, mA, lA, ogA, hi, qgA, key0);
    v16bf pbB = softmax_update(sB0, sB1, mB, lB, ogB, hi, qgB, key0);
    // ---- Oᵀ += Vᵀ·P : Vᵀ A-frags shared by both P fragments
#pragma unroll
    for (int g = 0; g < 4; ++g) {
      v16bf av = load_afrag(vb, SEQ, g * 16, key0, lane);
      ogA[g] = wmma_bf16(av, pbA, ogA[g]);
      ogB[g] = wmma_bf16(av, pbB, ogB[g]);
    }
  }

  // ---- normalize and scatter z as [b][s][h][dh] bf16
  float invA = 1.0f / lA, invB = 1.0f / lB;
  int b = bh >> 4, h = bh & 15;
#pragma unroll
  for (int g = 0; g < 4; ++g)
#pragma unroll
    for (int r = 0; r < 8; ++r) {
      int cc = g * 16 + hi * 8 + r;
      z_all[(((size_t)b * SEQ + qgA) * NHEADS + h) * DHEAD + cc] =
          f2bf(ogA[g][r] * invA);
      z_all[(((size_t)b * SEQ + qgB) * NHEADS + h) * DHEAD + cc] =
          f2bf(ogB[g][r] * invB);
    }
}

// ---------------------------------------------------------------------------
// Output projection: [8192,1024] x [1024,1024] + bias -> f32 out.
// Register-blocked 64x32 per wave (4x2 WMMA tiles).
// ---------------------------------------------------------------------------
__global__ __launch_bounds__(256) void k_outproj(const __bf16* __restrict__ z_all,
                                                 const __bf16* __restrict__ woutT,
                                                 const float* __restrict__ b_out,
                                                 float* __restrict__ out) {
  int wave = blockIdx.x * 8 + (threadIdx.x >> 5);
  int lane = threadIdx.x & 31;
  int rt = wave >> 5, ct = wave & 31;   // 128 x 32 tiles of 64x32

  v8f acc[4][2];
#pragma unroll
  for (int i = 0; i < 4; ++i)
#pragma unroll
    for (int j = 0; j < 2; ++j) acc[i][j] = vzero8();

#pragma unroll 2
  for (int k0 = 0; k0 < DMODEL; k0 += 32) {
    v16bf a[4], b[2];
#pragma unroll
    for (int i = 0; i < 4; ++i)
      a[i] = load_afrag(z_all, DMODEL, rt * 64 + i * 16, k0, lane);
#pragma unroll
    for (int j = 0; j < 2; ++j)
      b[j] = load_bfrag(woutT, DMODEL, ct * 32 + j * 16, k0, lane);
#pragma unroll
    for (int i = 0; i < 4; ++i)
#pragma unroll
      for (int j = 0; j < 2; ++j) acc[i][j] = wmma_bf16(a[i], b[j], acc[i][j]);
  }

  int hi = lane >> 4, n = lane & 15;
#pragma unroll
  for (int j = 0; j < 2; ++j) {
    int col = ct * 32 + j * 16 + n;
    float bias = b_out[col];
#pragma unroll
    for (int i = 0; i < 4; ++i)
#pragma unroll
      for (int r = 0; r < 8; ++r) {
        int row = rt * 64 + i * 16 + hi * 8 + r;
        out[(size_t)row * DMODEL + col] = acc[i][j][r] + bias;
      }
  }
}

// ---------------------------------------------------------------------------
extern "C" void kernel_launch(void* const* d_in, const int* in_sizes, int n_in,
                              void* d_out, int out_size, void* d_ws, size_t ws_size,
                              hipStream_t stream) {
  (void)in_sizes; (void)n_in; (void)out_size; (void)ws_size;
  const float* resid = (const float*)d_in[0];
  const float* W_Q   = (const float*)d_in[1];
  const float* W_K   = (const float*)d_in[2];
  const float* W_V   = (const float*)d_in[3];
  const float* W_out = (const float*)d_in[4];
  const float* b_out = (const float*)d_in[5];
  float* out = (float*)d_out;

  char* ws = (char*)d_ws;
  size_t off = 0;
  auto alloc = [&](size_t bytes) -> void* {
    void* p = ws + off;
    off += (bytes + 255) & ~(size_t)255;
    return p;
  };
  __bf16* resid_bf = (__bf16*)alloc((size_t)BS * DMODEL * 2);       // 16 MB
  __bf16* wqt      = (__bf16*)alloc((size_t)DMODEL * DMODEL * 2);   //  2 MB
  __bf16* wkt      = (__bf16*)alloc((size_t)DMODEL * DMODEL * 2);
  __bf16* wvt      = (__bf16*)alloc((size_t)DMODEL * DMODEL * 2);
  __bf16* woutT    = (__bf16*)alloc((size_t)DMODEL * DMODEL * 2);
  __bf16* q_all    = (__bf16*)alloc((size_t)BH * SEQ * DHEAD * 2);  // 16 MB
  __bf16* k_all    = (__bf16*)alloc((size_t)BH * SEQ * DHEAD * 2);
  __bf16* vT_all   = (__bf16*)alloc((size_t)BH * DHEAD * SEQ * 2);
  __bf16* z_all    = (__bf16*)alloc((size_t)BS * DMODEL * 2);       // 16 MB

  // prep
  k_conv_bf16<<<4096, 256, 0, stream>>>(resid, resid_bf, BS * DMODEL);
  k_transpose_w<<<4096, 256, 0, stream>>>(W_Q, wqt);
  k_transpose_w<<<4096, 256, 0, stream>>>(W_K, wkt);
  k_transpose_w<<<4096, 256, 0, stream>>>(W_V, wvt);
  k_prep_wout<<<4096, 256, 0, stream>>>(W_out, woutT);

  // QKV projection: 3 * 128 * 32 waves / 8 per block
  k_qkv<<<1536, 256, 0, stream>>>(resid_bf, wqt, wkt, wvt, q_all, k_all, vT_all);

  // flash attention: 64 (b,h) * 64 query blocks / 8 waves per block
  k_attn<<<512, 256, 0, stream>>>(q_all, k_all, vT_all, z_all);

  // output projection: 128 * 32 waves / 8 per block
  k_outproj<<<512, 256, 0, stream>>>(z_all, woutT, b_out, out);
}
